// HGNN_79551384256877
// MI455X (gfx1250) — matrix operand order repaired and checked by reading, not compiled, not run on hardware
//
#include <hip/hip_runtime.h>

#define HID 128
#define OUTDIM 64
#define NA 100000
#define NRS 20000
#define NRD 50000
#define EF  1600000
#define ERS 800000
#define ERD 800000
#define EHRD 800000

typedef __attribute__((ext_vector_type(16))) __bf16 v16bf;
typedef __attribute__((ext_vector_type(8)))  float v8f;

// ---------------------------------------------------------------------------
// Pre-pack W[128,128] (f32) into bf16 hi/lo buffers in WMMA B-fragment order:
//   index = ((ks*8 + nt)*32 + lane)*16 + e
//   where k = ks*32 + (lane>=16)*8 + (e&7) + (e>=8)*16,  col = nt*16 + (lane&15)
// Each lane of the GEMM then loads one fragment as 32 contiguous bytes.
// ---------------------------------------------------------------------------
__global__ void k_packW(const float* __restrict__ W, __bf16* __restrict__ Whi,
                        __bf16* __restrict__ Wlo)
{
  const int i = blockIdx.x * blockDim.x + threadIdx.x;
  if (i >= 32 * 32 * 16) return;              // 4ks * 8nt * 32lanes * 16e = 16384
  const int e    = i & 15;
  const int lane = (i >> 4) & 31;
  const int frag = i >> 9;                    // ks*8 + nt
  const int ks = frag >> 3, nt = frag & 7;
  const int k   = ks * 32 + (lane >> 4) * 8 + (e & 7) + (e >> 3) * 16;
  const int col = nt * 16 + (lane & 15);
  const float f = W[(size_t)k * HID + col];
  const __bf16 h = (__bf16)f;
  Whi[i] = h;
  Wlo[i] = (__bf16)(f - (float)h);
}

// ---------------------------------------------------------------------------
// Core 16x128 GEMM tile: one wave computes rows [tile*16, tile*16+16) of
// X[M,128] @ W[128,128] using bf16 split-precision WMMA (3 wmma per K-step).
// B comes pre-split/pre-swizzled from k_packW.
// ---------------------------------------------------------------------------
static __device__ __forceinline__ void gemm_tile(const float* __restrict__ X,
                                                 const __bf16* __restrict__ Whi,
                                                 const __bf16* __restrict__ Wlo,
                                                 int tile, int lane, v8f* acc)
{
  const int rlane = lane & 15;
  const int khalf = lane >> 4;
  const float* xrow = X + (size_t)(tile * 16 + rlane) * HID;

#pragma unroll
  for (int ks = 0; ks < 4; ++ks) {
    // A (16x32 bf16): lanes 0-15 hold K=[ks*32+0..7, +16..23]; lanes 16-31 shift +8
    const float* xp = xrow + ks * 32 + khalf * 8;
    v16bf Ahi, Alo;
#pragma unroll
    for (int e = 0; e < 8; ++e) {
      float f = xp[e];
      __bf16 h = (__bf16)f;
      Ahi[e] = h; Alo[e] = (__bf16)(f - (float)h);
      float g = xp[16 + e];
      __bf16 hg = (__bf16)g;
      Ahi[8 + e] = hg; Alo[8 + e] = (__bf16)(g - (float)hg);
    }
#pragma unroll
    for (int nt = 0; nt < 8; ++nt) {
      const size_t idx = ((size_t)(ks * 8 + nt) * 32 + lane) * 16;
      const v16bf Bhi = *(const v16bf*)(Whi + idx);
      const v16bf Blo = *(const v16bf*)(Wlo + idx);
      acc[nt] = __builtin_amdgcn_wmma_f32_16x16x32_bf16(false, Ahi, false, Bhi,
                                                        (short)0, acc[nt], false, false);
      acc[nt] = __builtin_amdgcn_wmma_f32_16x16x32_bf16(false, Ahi, false, Blo,
                                                        (short)0, acc[nt], false, false);
      acc[nt] = __builtin_amdgcn_wmma_f32_16x16x32_bf16(false, Alo, false, Bhi,
                                                        (short)0, acc[nt], false, false);
    }
  }
}

// hs = X @ W, stored to H (M multiple of 16; grid = M/16, block = 32)
__global__ __launch_bounds__(32) void k_gemm128(const float* __restrict__ X,
                                                const __bf16* __restrict__ Whi,
                                                const __bf16* __restrict__ Wlo,
                                                float* __restrict__ H)
{
  const int tile = blockIdx.x;
  const int lane = threadIdx.x & 31;
  const int rlane = lane & 15;
  const int khalf = lane >> 4;
  v8f acc[8] = {};
  gemm_tile(X, Whi, Wlo, tile, lane, acc);
  // C layout: VGPR j -> row j + 8*khalf; lane -> col nt*16+rlane
  float* hbase = H + (size_t)(tile * 16 + khalf * 8) * HID;
#pragma unroll
  for (int nt = 0; nt < 8; ++nt)
#pragma unroll
    for (int j = 0; j < 8; ++j)
      hbase[(size_t)j * HID + nt * 16 + rlane] = acc[nt][j];
}

// feat[col] += sum_rows relu( (X@W)[row,col] + b[col] )   (readout column-sums)
__global__ __launch_bounds__(32) void k_meanln(const float* __restrict__ X,
                                               const __bf16* __restrict__ Whi,
                                               const __bf16* __restrict__ Wlo,
                                               const float* __restrict__ b,
                                               float* __restrict__ feat)
{
  const int tile = blockIdx.x;
  const int lane = threadIdx.x & 31;
  const int rlane = lane & 15;
  v8f acc[8] = {};
  gemm_tile(X, Whi, Wlo, tile, lane, acc);
#pragma unroll
  for (int nt = 0; nt < 8; ++nt) {
    const int col = nt * 16 + rlane;
    const float bc = b[col];
    float s = 0.f;
#pragma unroll
    for (int j = 0; j < 8; ++j) s += fmaxf(acc[nt][j] + bc, 0.f);
    atomicAdd(feat + col, s);
  }
}

// v[k] = sum_n W[k,n] * a[n]   (fold Wdst @ att_dst / Wsrc @ att_src)
__global__ void k_matvec(const float* __restrict__ W, const float* __restrict__ a,
                         float* __restrict__ v)
{
  const int k = threadIdx.x;
  float s = 0.f;
  for (int n = 0; n < HID; ++n) s += W[(size_t)k * HID + n] * a[n];
  v[k] = s;
}

// o[n] = x[n,:] . v   (wave per node)
__global__ void k_dotn(const float* __restrict__ x, const float* __restrict__ v,
                       float* __restrict__ o, int N)
{
  const int w = (int)((blockIdx.x * blockDim.x + threadIdx.x) >> 5);
  const int lane = threadIdx.x & 31;
  if (w >= N) return;
  const float* xr = x + (size_t)w * HID + lane * 4;
  const float* vr = v + lane * 4;
  float s = xr[0] * vr[0] + xr[1] * vr[1] + xr[2] * vr[2] + xr[3] * vr[3];
  for (int off = 16; off > 0; off >>= 1) s += __shfl_down(s, off, 32);
  if (lane == 0) o[w] = s;
}

__global__ void k_initbias(float* __restrict__ o, const float* __restrict__ b1,
                           const float* __restrict__ b2, int n)
{
  const int i = blockIdx.x * blockDim.x + threadIdx.x;
  if (i >= n) return;
  const int c = i & (HID - 1);
  float v = b1[c];
  if (b2) v += b2[c];
  o[i] = v;
}

__global__ void k_relu(float* __restrict__ o, int n)
{
  const int i = blockIdx.x * blockDim.x + threadIdx.x;
  if (i < n) o[i] = fmaxf(o[i], 0.f);
}

static __device__ __forceinline__ unsigned enc_f(float f)
{
  unsigned u = __float_as_uint(f);
  return (u & 0x80000000u) ? ~u : (u | 0x80000000u);
}
static __device__ __forceinline__ float dec_f(unsigned u)
{
  u = (u & 0x80000000u) ? (u ^ 0x80000000u) : ~u;
  return __uint_as_float(u);
}

// pass A: e = leaky_relu(as[src]+ad[dst]); running segment max (encoded atomicMax)
__global__ void k_edgeA(const int* __restrict__ src, const int* __restrict__ dst,
                        const float* __restrict__ as_, const float* __restrict__ ad_,
                        float* __restrict__ eb, unsigned* __restrict__ menc, int E)
{
  const int i = blockIdx.x * blockDim.x + threadIdx.x;
  if (i >= E) return;
  float v = as_[src[i]] + ad_[dst[i]];
  v = (v > 0.f) ? v : 0.2f * v;
  eb[i] = v;
  atomicMax(menc + dst[i], enc_f(v));
}

// pass B: p = exp(e - m[dst]); z[dst] += p
__global__ void k_edgeB(const int* __restrict__ dst, const float* __restrict__ eb,
                        const unsigned* __restrict__ menc, float* __restrict__ pb,
                        float* __restrict__ zb, int E)
{
  const int i = blockIdx.x * blockDim.x + threadIdx.x;
  if (i >= E) return;
  const int d = dst[i];
  const float pp = __expf(eb[i] - dec_f(menc[d]));
  pb[i] = pp;
  atomicAdd(zb + d, pp);
}

// pass C: out[dst,:] += (p/z[dst]) * hs[src,:]   (wave per edge, 4 floats/lane)
__global__ void k_edgeC(const int* __restrict__ src, const int* __restrict__ dst,
                        const float* __restrict__ pb, const float* __restrict__ zb,
                        const float* __restrict__ hs, float* __restrict__ out, int E)
{
  const int w = (int)((blockIdx.x * blockDim.x + threadIdx.x) >> 5);
  const int lane = threadIdx.x & 31;
  if (w >= E) return;
  const int s = src[w], d = dst[w];
  const float wt = pb[w] / fmaxf(zb[d], 1e-16f);
  const float* hr = hs + (size_t)s * HID + lane * 4;
  float* orow = out + (size_t)d * HID + lane * 4;
#pragma unroll
  for (int c = 0; c < 4; ++c) atomicAdd(orow + c, wt * hr[c]);
}

// out[t] = b_fc[t] + pooled . W_fc[:,t]
__global__ void k_final(const float* __restrict__ feat, const float* __restrict__ Wfc,
                        const float* __restrict__ bfc, float* __restrict__ out)
{
  const int t = threadIdx.x;
  if (t >= OUTDIM) return;
  float s = bfc[t];
  for (int k = 0; k < HID; ++k) {
    float pk = (feat[k] * (1.0f / NA) + feat[HID + k] * (1.0f / NRS) +
                feat[2 * HID + k] * (1.0f / NRD)) * (1.0f / 3.0f);
    s += pk * Wfc[(size_t)k * OUTDIM + t];
  }
  out[t] = s;
}

extern "C" void kernel_launch(void* const* d_in, const int* in_sizes, int n_in,
                              void* d_out, int out_size, void* d_ws, size_t ws_size,
                              hipStream_t stream)
{
  (void)in_sizes; (void)n_in; (void)out_size; (void)ws_size;
  const float* x_act = (const float*)d_in[0];
  const float* x_rs  = (const float*)d_in[1];
  const float* x_rd  = (const float*)d_in[2];
  const int* ef_src  = (const int*)d_in[3];
  const int* ef_dst  = (const int*)d_in[4];
  const int* ers_src = (const int*)d_in[5];
  const int* ers_dst = (const int*)d_in[6];
  const int* erd_src = (const int*)d_in[7];
  const int* erd_dst = (const int*)d_in[8];
  const int* ehrd_src = (const int*)d_in[9];
  const int* ehrd_dst = (const int*)d_in[10];
  const float* Wsrc    = (const float*)d_in[11];
  const float* Wdst    = (const float*)d_in[12];
  const float* att_src = (const float*)d_in[13];
  const float* att_dst = (const float*)d_in[14];
  const float* b_gat   = (const float*)d_in[15];
  const float* W_ln = (const float*)d_in[16];
  const float* b_ln = (const float*)d_in[17];
  const float* W_fc = (const float*)d_in[18];
  const float* b_fc = (const float*)d_in[19];

  float* w = (float*)d_ws;
  size_t off = 0;
  auto alloc = [&](size_t n) { float* p = w + off; off += (n + 63) & ~(size_t)63; return p; };
  float* xa0  = alloc((size_t)NA * HID);
  float* xa1  = alloc((size_t)NA * HID);
  float* xrs0 = alloc((size_t)NRS * HID);
  float* xrs1 = alloc((size_t)NRS * HID);
  float* xrd0 = alloc((size_t)NRD * HID);
  float* xrd1 = alloc((size_t)NRD * HID);
  float* hs   = alloc((size_t)NA * HID);
  float* asb  = alloc(NA);
  float* adb  = alloc(NA);
  float* vs   = alloc(HID);
  float* vd   = alloc(HID);
  unsigned* menc = (unsigned*)alloc(NA);
  float* zb   = alloc(NA);
  float* eb   = alloc(EF);
  float* pb   = alloc(EF);
  float* feat = alloc(3 * HID);
  __bf16* whi = (__bf16*)alloc(HID * HID / 2);   // 16384 bf16
  __bf16* wlo = (__bf16*)alloc(HID * HID / 2);

  const int PACK_N = HID * HID;                  // 16384 pack work-items

  for (int L = 0; L < 2; ++L) {
    const float* xa  = L ? xa0  : x_act;
    const float* xrs = L ? xrs0 : x_rs;
    const float* xrd = L ? xrd0 : x_rd;
    float* na  = L ? xa1  : xa0;
    float* nrs = L ? xrs1 : xrs0;
    float* nrd = L ? xrd1 : xrd0;

    k_initbias<<<(NA * HID + 255) / 256, 256, 0, stream>>>(na,  b_gat + (0 * 2 + L) * HID, nullptr, NA * HID);
    k_initbias<<<(NRS * HID + 255) / 256, 256, 0, stream>>>(nrs, b_gat + (1 * 2 + L) * HID, nullptr, NRS * HID);
    k_initbias<<<(NRD * HID + 255) / 256, 256, 0, stream>>>(nrd, b_gat + (2 * 2 + L) * HID,
                                                            b_gat + (3 * 2 + L) * HID, NRD * HID);

    struct RelDesc { const float* xs; int ns; const float* xd; int nd;
                     const int* s; const int* d; int E; float* out; };
    RelDesc rel[4] = {
      { xa,  NA,  xa,  NA,  ef_src,   ef_dst,   EF,   na  },
      { xa,  NA,  xrs, NRS, ers_src,  ers_dst,  ERS,  nrs },
      { xrd, NRD, xrd, NRD, erd_src,  erd_dst,  ERD,  nrd },
      { xa,  NA,  xrd, NRD, ehrd_src, ehrd_dst, EHRD, nrd },
    };
    for (int r = 0; r < 4; ++r) {
      const RelDesc& R = rel[r];
      const float* Ws = Wsrc + (size_t)(r * 2 + L) * HID * HID;
      const float* Wd = Wdst + (size_t)(r * 2 + L) * HID * HID;
      k_matvec<<<1, HID, 0, stream>>>(Ws, att_src + (r * 2 + L) * HID, vs);
      k_matvec<<<1, HID, 0, stream>>>(Wd, att_dst + (r * 2 + L) * HID, vd);
      k_packW<<<(PACK_N + 255) / 256, 256, 0, stream>>>(Ws, whi, wlo);
      k_gemm128<<<R.ns / 16, 32, 0, stream>>>(R.xs, whi, wlo, hs);
      k_dotn<<<(R.ns * 32 + 255) / 256, 256, 0, stream>>>(R.xs, vs, asb, R.ns);
      k_dotn<<<(R.nd * 32 + 255) / 256, 256, 0, stream>>>(R.xd, vd, adb, R.nd);
      hipMemsetAsync(menc, 0, (size_t)R.nd * 4, stream);
      hipMemsetAsync(zb,   0, (size_t)R.nd * 4, stream);
      k_edgeA<<<(R.E + 255) / 256, 256, 0, stream>>>(R.s, R.d, asb, adb, eb, menc, R.E);
      k_edgeB<<<(R.E + 255) / 256, 256, 0, stream>>>(R.d, eb, menc, pb, zb, R.E);
      k_edgeC<<<(unsigned)(((size_t)R.E * 32 + 255) / 256), 256, 0, stream>>>(R.s, R.d, pb, zb, hs, R.out, R.E);
    }
    k_relu<<<(NA * HID + 255) / 256, 256, 0, stream>>>(na,  NA * HID);
    k_relu<<<(NRS * HID + 255) / 256, 256, 0, stream>>>(nrs, NRS * HID);
    k_relu<<<(NRD * HID + 255) / 256, 256, 0, stream>>>(nrd, NRD * HID);
  }

  hipMemsetAsync(feat, 0, 3 * HID * 4, stream);
  k_packW<<<(PACK_N + 255) / 256, 256, 0, stream>>>(W_ln, whi, wlo);
  k_meanln<<<NA / 16, 32, 0, stream>>>(xa1, whi, wlo, b_ln, feat);
  k_meanln<<<NRS / 16, 32, 0, stream>>>(xrs1, whi, wlo, b_ln, feat + HID);
  k_meanln<<<NRD / 16, 32, 0, stream>>>(xrd1, whi, wlo, b_ln, feat + 2 * HID);
  k_final<<<1, 64, 0, stream>>>(feat, W_fc, b_fc, (float*)d_out);
}
